// T2TAttention_47304769798207
// MI455X (gfx1250) — matrix operand order repaired and checked by reading, not compiled, not run on hardware
//
#include <hip/hip_runtime.h>

typedef __attribute__((ext_vector_type(16))) _Float16 v16h;
typedef __attribute__((ext_vector_type(8)))  _Float16 v8h;
typedef __attribute__((ext_vector_type(8)))  float    v8f;

#define BATCH 8
#define SEQ   3136
#define DIM   64
#define ROWS  (BATCH*SEQ)

#define WMMA_F16(A,B,C) __builtin_amdgcn_wmma_f32_16x16x32_f16(false,(A),false,(B),(short)0,(C),false,false)

// ---- fragment loaders (wave32 WMMA; this code's k order k = 16*half16 + s for A and B, not the ISA table order) ----

// A 16x32 f16: lane16 = row M; lanes 0-15 hold K 0..15, lanes 16-31 hold K 16..31
// Two aligned 16-byte chunks per lane -> b128 loads.
__device__ __forceinline__ v16h load_A_h16(const _Float16* base, int ldh, int lane16, int half16) {
  const _Float16* r = base + (size_t)lane16 * ldh + half16*16;
  v8h lo = *(const v8h*)(r);
  v8h hi = *(const v8h*)(r + 8);
  v16h a;
  #pragma unroll
  for (int e = 0; e < 8; ++e) { a[e] = lo[e]; a[8+e] = hi[e]; }
  return a;
}

__device__ __forceinline__ v16h load_A_f32h(const float* base, int ld, int lane16, int half16) {
  v16h a;
  const float* r = base + (size_t)lane16 * ld;
  #pragma unroll
  for (int e = 0; e < 8; ++e) a[e]     = (_Float16)r[half16*16 + e];
  #pragma unroll
  for (int e = 0; e < 8; ++e) a[8 + e] = (_Float16)r[half16*16 + 8 + e];
  return a;
}

// B 32x16 f16 (K x N) from an N-major source src[n*ldh + k]:
// lane16 = N, half16 selects K 0-15 / 16-31 -> one aligned 32-byte load per lane.
__device__ __forceinline__ v16h load_B_nmajor(const _Float16* base, int ldh, int lane16, int half16) {
  const _Float16* p = base + (size_t)lane16 * ldh + half16*16;
  return *(const v16h*)p;
}

// B 32x16 from a K-major f32 source src[k*ld + n] (small weight matrices, L2-resident)
__device__ __forceinline__ v16h load_B_kmajor_f32(const float* base, int ld, int lane16, int half16) {
  v16h b;
  #pragma unroll
  for (int e = 0; e < 16; ++e) b[e] = (_Float16)base[(size_t)(half16*16 + e)*ld + lane16];
  return b;
}

// ---- kernel 1: qkv = x @ W_qkv; emit q16 (scaled), k16, vT16 (transposed), v32 ----
__global__ __launch_bounds__(256) void qkv_kernel(const float* __restrict__ x,
    const float* __restrict__ Wqkv, _Float16* __restrict__ q16,
    _Float16* __restrict__ k16, _Float16* __restrict__ vT16,
    float* __restrict__ v32)
{
  int wave   = blockIdx.x*8 + (threadIdx.x >> 5);
  int lane   = threadIdx.x & 31;
  int lane16 = lane & 15, half16 = lane >> 4;
  int row0   = wave * 16;

  v16h A0 = load_A_f32h(x + (size_t)row0*DIM,      DIM, lane16, half16);
  v16h A1 = load_A_f32h(x + (size_t)row0*DIM + 32, DIM, lane16, half16);

  #pragma unroll
  for (int nt = 0; nt < 12; ++nt) {
    int n0 = nt * 16;
    v8f c = {};
    v16h b0 = load_B_kmajor_f32(Wqkv + n0,            192, lane16, half16);
    c = WMMA_F16(A0, b0, c);
    v16h b1 = load_B_kmajor_f32(Wqkv + 32*192 + n0,   192, lane16, half16);
    c = WMMA_F16(A1, b1, c);
    #pragma unroll
    for (int r = 0; r < 8; ++r) {
      int row = row0 + r + half16*8;
      if (n0 < 64) {                 // q, pre-scaled by 1/sqrt(64)
        q16[(size_t)row*DIM + n0 + lane16] = (_Float16)(c[r] * 0.125f);
      } else if (n0 < 128) {         // k
        k16[(size_t)row*DIM + (n0 - 64) + lane16] = (_Float16)c[r];
      } else {                       // v: f32 residual copy + f16 transposed copy
        int d = n0 - 128 + lane16;
        v32[(size_t)row*DIM + d] = c[r];
        int bi = row / SEQ, tok = row % SEQ;
        vT16[((size_t)bi*DIM + d)*SEQ + tok] = (_Float16)c[r];
      }
    }
  }
}

// ---- kernel 2: flash attention, one wave per 32-query tile ----
// 32 query rows per wave: each K/V B-fragment load feeds two WMMAs (2x FLOPs/byte).
__global__ __launch_bounds__(256) void attn_kernel(const _Float16* __restrict__ q16,
    const _Float16* __restrict__ k16, const _Float16* __restrict__ vT16,
    float* __restrict__ attn32)
{
  __shared__ _Float16 ldsP[8][32*32];   // per-wave 32x32 P tile (C-layout -> A-layout)
  int wwave  = threadIdx.x >> 5;
  int wave   = blockIdx.x*8 + wwave;
  int lane   = threadIdx.x & 31;
  int lane16 = lane & 15, half16 = lane >> 4;
  int bi     = wave / (SEQ/32);
  int t0     = (wave % (SEQ/32)) * 32;
  size_t row0 = (size_t)bi*SEQ + t0;

  const _Float16* kb  = k16  + (size_t)bi*SEQ*DIM;
  const _Float16* vtb = vT16 + (size_t)bi*DIM*SEQ;

  v16h Q[2][2];
  #pragma unroll
  for (int u = 0; u < 2; ++u) {
    Q[u][0] = load_A_h16(q16 + (row0 + u*16)*DIM,      DIM, lane16, half16);
    Q[u][1] = load_A_h16(q16 + (row0 + u*16)*DIM + 32, DIM, lane16, half16);
  }

  v8f O[2][4] = {};
  float m[2][8], l[2][8];
  #pragma unroll
  for (int u = 0; u < 2; ++u)
    #pragma unroll
    for (int r = 0; r < 8; ++r) { m[u][r] = -1e30f; l[u][r] = 0.f; }

  _Float16* P = ldsP[wwave];

  for (int j0 = 0; j0 < SEQ; j0 += 32) {
    // S = Q @ K^T for 32 keys x 32 query rows (4 C tiles, 8 WMMAs, 4 K-fragment loads)
    v8f S[2][2] = {};
    #pragma unroll
    for (int kt = 0; kt < 2; ++kt) {
      v16h bk0 = load_B_nmajor(kb + (size_t)(j0 + 16*kt)*DIM,      DIM, lane16, half16);
      v16h bk1 = load_B_nmajor(kb + (size_t)(j0 + 16*kt)*DIM + 32, DIM, lane16, half16);
      #pragma unroll
      for (int u = 0; u < 2; ++u) {
        S[u][kt] = WMMA_F16(Q[u][0], bk0, S[u][kt]);
        S[u][kt] = WMMA_F16(Q[u][1], bk1, S[u][kt]);
      }
    }
    // block row-max over 32 keys: combine column tiles, butterfly over 16-lane column group
    float bm[2][8];
    #pragma unroll
    for (int u = 0; u < 2; ++u)
      #pragma unroll
      for (int r = 0; r < 8; ++r) bm[u][r] = fmaxf(S[u][0][r], S[u][1][r]);
    #pragma unroll
    for (int off = 1; off < 16; off <<= 1)
      #pragma unroll
      for (int u = 0; u < 2; ++u)
        #pragma unroll
        for (int r = 0; r < 8; ++r) bm[u][r] = fmaxf(bm[u][r], __shfl_xor(bm[u][r], off, 32));

    // online softmax update; write 32x32 P tile (f16) to LDS row-major
    float al[2][8], rs[2][8];
    #pragma unroll
    for (int u = 0; u < 2; ++u)
      #pragma unroll
      for (int r = 0; r < 8; ++r) {
        float mn = fmaxf(m[u][r], bm[u][r]);
        al[u][r] = __expf(m[u][r] - mn);
        m[u][r]  = mn;
        float p0 = __expf(S[u][0][r] - mn);
        float p1 = __expf(S[u][1][r] - mn);
        int row = u*16 + r + half16*8;
        P[row*32 + lane16]      = (_Float16)p0;
        P[row*32 + 16 + lane16] = (_Float16)p1;
        rs[u][r] = p0 + p1;
      }
    #pragma unroll
    for (int off = 1; off < 16; off <<= 1)
      #pragma unroll
      for (int u = 0; u < 2; ++u)
        #pragma unroll
        for (int r = 0; r < 8; ++r) rs[u][r] += __shfl_xor(rs[u][r], off, 32);
    #pragma unroll
    for (int u = 0; u < 2; ++u)
      #pragma unroll
      for (int r = 0; r < 8; ++r) l[u][r] = l[u][r]*al[u][r] + rs[u][r];

    // LDS stores -> loads within this wave: DS is in-order, make it explicit
    asm volatile("s_wait_dscnt 0" ::: "memory");
    v16h Pa[2];
    Pa[0] = load_A_h16(P,         32, lane16, half16);
    Pa[1] = load_A_h16(P + 16*32, 32, lane16, half16);

    // O = O*alpha + P @ V  (4 V-fragment loads feed 8 WMMAs)
    #pragma unroll
    for (int u = 0; u < 2; ++u)
      #pragma unroll
      for (int dt = 0; dt < 4; ++dt)
        #pragma unroll
        for (int r = 0; r < 8; ++r) O[u][dt][r] *= al[u][r];
    #pragma unroll
    for (int dt = 0; dt < 4; ++dt) {
      v16h bv = load_B_nmajor(vtb + (size_t)(dt*16)*SEQ + j0, SEQ, lane16, half16);
      #pragma unroll
      for (int u = 0; u < 2; ++u) O[u][dt] = WMMA_F16(Pa[u], bv, O[u][dt]);
    }
  }

  #pragma unroll
  for (int u = 0; u < 2; ++u)
    #pragma unroll
    for (int r = 0; r < 8; ++r) {
      size_t row = row0 + u*16 + r + half16*8;
      float inv = 1.f / l[u][r];
      #pragma unroll
      for (int dt = 0; dt < 4; ++dt)
        attn32[row*DIM + dt*16 + lane16] = O[u][dt][r]*inv;
    }
}

// ---- kernel 3: out = attn @ W_proj + b_proj + v ----
__global__ __launch_bounds__(256) void proj_kernel(const float* __restrict__ attn32,
    const float* __restrict__ Wp, const float* __restrict__ bp,
    const float* __restrict__ v32, float* __restrict__ out)
{
  int wave   = blockIdx.x*8 + (threadIdx.x >> 5);
  int lane   = threadIdx.x & 31;
  int lane16 = lane & 15, half16 = lane >> 4;
  size_t row0 = (size_t)wave * 16;

  v16h A0 = load_A_f32h(attn32 + row0*DIM,      DIM, lane16, half16);
  v16h A1 = load_A_f32h(attn32 + row0*DIM + 32, DIM, lane16, half16);

  #pragma unroll
  for (int nt = 0; nt < 4; ++nt) {
    int n0 = nt * 16;
    v8f c = {};
    v16h b0 = load_B_kmajor_f32(Wp + n0,          DIM, lane16, half16);
    c = WMMA_F16(A0, b0, c);
    v16h b1 = load_B_kmajor_f32(Wp + 32*DIM + n0, DIM, lane16, half16);
    c = WMMA_F16(A1, b1, c);
    float bias = bp[n0 + lane16];
    #pragma unroll
    for (int r = 0; r < 8; ++r) {
      size_t row = row0 + r + half16*8;
      out[row*DIM + n0 + lane16] = c[r] + bias + v32[row*DIM + n0 + lane16];
    }
  }
}

extern "C" void kernel_launch(void* const* d_in, const int* in_sizes, int n_in,
                              void* d_out, int out_size, void* d_ws, size_t ws_size,
                              hipStream_t stream) {
  const float* x    = (const float*)d_in[0];
  const float* Wqkv = (const float*)d_in[1];
  const float* Wp   = (const float*)d_in[2];
  const float* bp   = (const float*)d_in[3];
  float* out = (float*)d_out;

  const size_t per = (size_t)ROWS * DIM;        // 1,605,632 elements
  char* ws = (char*)d_ws;
  _Float16* q16    = (_Float16*)(ws);
  _Float16* k16    = (_Float16*)(ws + per*2);
  _Float16* vT16   = (_Float16*)(ws + per*4);
  float*    v32    = (float*)   (ws + per*6);
  float*    attn32 = (float*)   (ws + per*6 + per*4);
  if (ws_size < per*14) return;                 // ~21.5 MB needed

  dim3 blk(256);                                // 8 waves of 32
  qkv_kernel <<<dim3(ROWS/16/8), blk, 0, stream>>>(x, Wqkv, q16, k16, vT16, v32);
  attn_kernel<<<dim3(ROWS/32/8), blk, 0, stream>>>(q16, k16, vT16, attn32);
  proj_kernel<<<dim3(ROWS/16/8), blk, 0, stream>>>(attn32, Wp, bp, v32, out);
}
